// Fpndecoder_47553877902043
// MI455X (gfx1250) — matrix operand order repaired and checked by reading, not compiled, not run on hardware
//
#include <hip/hip_runtime.h>
#include <hip/hip_bf16.h>
#include <math.h>

typedef _Float16 v16h __attribute__((ext_vector_type(16)));
typedef _Float16 v8h  __attribute__((ext_vector_type(8)));
typedef float    v8f  __attribute__((ext_vector_type(8)));
typedef int      i32x4 __attribute__((ext_vector_type(4)));

#define B_   8
#define C_   256
#define H_   128
#define W_   128
#define PH   32
#define PW   32
#define NTOK 1024        // PH*PW
#define NH   8
#define HD   32
#define FFN  1024
#define CPBH 512

#ifndef __has_builtin
#define __has_builtin(x) 0
#endif
#if __has_builtin(__builtin_amdgcn_global_load_async_to_lds_b128) && \
    __has_builtin(__builtin_amdgcn_s_wait_asynccnt)
#define HAVE_ASYNC_LDS 1
#else
#define HAVE_ASYNC_LDS 0
#endif

// builtin signature (from hipcc diagnostic): (v4i AS1*, v4i AS3*, imm offset, imm cpol)
#define GPTR128(p) ((__attribute__((address_space(1))) i32x4*)(void*)(p))
#define LPTR128(p) ((__attribute__((address_space(3))) i32x4*)(void*)(p))

static __device__ __forceinline__ v8f wmma16(v16h a, v16h b, v8f c) {
  // D = A(16x32 f16) * B(32x16 f16) + C(16x16 f32)
  return __builtin_amdgcn_wmma_f32_16x16x32_f16(false, a, false, b, (short)0, c, false, false);
}

// ---------------- weight conversion ----------------
__global__ void k_f32_to_f16(const float* __restrict__ src, _Float16* __restrict__ dst, int n) {
  int i = blockIdx.x * blockDim.x + threadIdx.x;
  if (i < n) dst[i] = (_Float16)src[i];
}

// ---------------- downsample 128->32 (+relu) to [B,N,C] f16 ----------------
// align_corners=False, scale=4: src coord = 4*i + 1.5 -> mean of 2x2 interior pixels
__global__ void k_downsample(const float* __restrict__ x, _Float16* __restrict__ out) {
  int i = blockIdx.x * blockDim.x + threadIdx.x;
  if (i >= B_ * C_ * PH * PW) return;
  int pw = i & 31; int t = i >> 5;
  int ph = t & 31; t >>= 5;
  int c  = t & 255; int b = t >> 8;
  const float* p = x + ((size_t)(b * C_ + c) * H_ + (4 * ph + 1)) * W_ + (4 * pw + 1);
  float v = 0.25f * (p[0] + p[1] + p[W_] + p[W_ + 1]);
  v = fmaxf(v, 0.0f);
  out[((size_t)b * NTOK + ph * PW + pw) * C_ + c] = (_Float16)v;
}

// ---------------- CPB MLP: [L,2] -> relu(512) -> [L,8] ----------------
__global__ void k_cpb(const float* __restrict__ table, const float* __restrict__ w1,
                      const float* __restrict__ b1, const float* __restrict__ w2,
                      const float* __restrict__ b2, float* __restrict__ cpb, int L) {
  int li = blockIdx.x * blockDim.x + threadIdx.x;
  if (li >= L) return;
  float t0 = table[2 * li], t1 = table[2 * li + 1];
  float acc[NH];
#pragma unroll
  for (int hh = 0; hh < NH; ++hh) acc[hh] = b2[hh];
  for (int j = 0; j < CPBH; ++j) {
    float hv = fmaxf(t0 * w1[j] + t1 * w1[CPBH + j] + b1[j], 0.0f);
#pragma unroll
    for (int hh = 0; hh < NH; ++hh) acc[hh] += hv * w2[j * NH + hh];
  }
#pragma unroll
  for (int hh = 0; hh < NH; ++hh) cpb[li * NH + hh] = acc[hh];
}

// ---------------- WMMA GEMM: 4 waves/block, shared B panel in LDS ----------------
// Block covers 64 rows (4 M-tiles, one per wave) x one 16-col N-tile.
// B k-panel [K,16] staged once into LDS via async global->LDS copies (ASYNCcnt).
enum { EP_F16 = 0, EP_QSCAT = 1, EP_KVSCAT = 2, EP_F32_AND_F16 = 3, EP_GELU_F16 = 4, EP_ADD_F32 = 5 };

__global__ __launch_bounds__(128)
void k_gemm16(const _Float16* __restrict__ A, const _Float16* __restrict__ Bm,
              const float* __restrict__ bias, int M, int K, int Nout, int mode,
              _Float16* __restrict__ o16a, _Float16* __restrict__ o16b,
              float* __restrict__ of32) {
  __shared__ _Float16 bpanel[FFN * 16];     // max K=1024 -> 32 KB
  int tid = threadIdx.x;
  int lane = tid & 31;
  int wv = tid >> 5;
  int ntiles = Nout >> 4;
  int bm = blockIdx.x / ntiles;
  int tn = blockIdx.x % ntiles;

  // ---- stage B panel: rows kk of Bm, columns [tn*16, tn*16+16) ----
  int nchunks = K * 2;                      // 16-byte chunks, 2 per row
  for (int ch = tid; ch < nchunks; ch += 128) {
    int kk = ch >> 1, half = (ch & 1) * 8;
    const _Float16* src = Bm + (size_t)kk * Nout + tn * 16 + half;
    _Float16* dst = &bpanel[kk * 16 + half];
#if HAVE_ASYNC_LDS
    __builtin_amdgcn_global_load_async_to_lds_b128(GPTR128(src), LPTR128(dst), 0, 0);
#else
    *(uint4*)dst = *(const uint4*)src;
#endif
  }
#if HAVE_ASYNC_LDS
  __builtin_amdgcn_s_wait_asynccnt(0);
#endif
  __syncthreads();

  int tm = bm * 4 + wv;
  int row = tm * 16 + (lane & 15);
  int khalf = ((lane >> 4) & 1) * 8;        // A layout: lanes 16-31 hold K=8..15 / 24..31
  v8f c = {};
  for (int kk = 0; kk < K; kk += 32) {
    const _Float16* ap = A + (size_t)row * K + kk + khalf;
    v8h lo = *(const v8h*)(ap);
    v8h hi = *(const v8h*)(ap + 16);
    v16h a;
#pragma unroll
    for (int i = 0; i < 8; ++i) { a[i] = lo[i]; a[i + 8] = hi[i]; }
    // B layout: lane%16 = N col, elements = K rows khalf+0..7 / khalf+16..23 (from LDS)
    v16h b;
#pragma unroll
    for (int i = 0; i < 8; ++i) {
      b[i]     = bpanel[(kk + khalf + i) * 16 + (lane & 15)];
      b[i + 8] = bpanel[(kk + khalf + 16 + i) * 16 + (lane & 15)];
    }
    c = wmma16(a, b, c);
  }
  int n = lane & 15;
  int mbase = (lane >> 4) * 8;              // C/D layout: lanes16-31 hold M=8..15
#pragma unroll
  for (int r = 0; r < 8; ++r) {
    int m = tm * 16 + mbase + r;
    int col = tn * 16 + n;
    float v = c[r] + bias[col];
    if (mode == EP_F16) {
      o16a[(size_t)m * Nout + col] = (_Float16)v;
    } else if (mode == EP_QSCAT) {          // -> Qh [B,h,N,d] f16
      int bb = m >> 10, nn = m & 1023;
      int hh = col >> 5, dd = col & 31;
      o16a[(((size_t)(bb * NH + hh) * NTOK + nn) << 5) + dd] = (_Float16)v;
    } else if (mode == EP_KVSCAT) {         // K -> [B,h,d,N], V -> [B,h,N,d]
      int bb = m >> 10, nn = m & 1023;
      if (col < 256) {
        int hh = col >> 5, dd = col & 31;
        o16a[((size_t)(bb * NH + hh) * HD + dd) * NTOK + nn] = (_Float16)v;
      } else {
        int c2 = col - 256;
        int hh = c2 >> 5, dd = c2 & 31;
        o16b[(((size_t)(bb * NH + hh) * NTOK + nn) << 5) + dd] = (_Float16)v;
      }
    } else if (mode == EP_F32_AND_F16) {    // Wo: f32 residual path + f16 FFN input
      of32[(size_t)m * Nout + col] = v;
      o16a[(size_t)m * Nout + col] = (_Float16)v;
    } else if (mode == EP_GELU_F16) {       // exact gelu
      float g = 0.5f * v * (1.0f + erff(v * 0.70710678118654752f));
      o16a[(size_t)m * Nout + col] = (_Float16)g;
    } else {                                // EP_ADD_F32: z += ffn2
      of32[(size_t)m * Nout + col] += v;
    }
  }
}

// ---------------- attention: 1 wave per (b, head, 16-query block) ----------------
__global__ __launch_bounds__(32)
void k_attention(const _Float16* __restrict__ Qh, const _Float16* __restrict__ Kt,
                 const _Float16* __restrict__ Vh, const int* __restrict__ relidx,
                 const float* __restrict__ cpb, _Float16* __restrict__ O16) {
  __shared__ float smem[16 * NTOK];      // 64 KB logits strip
  int lane = threadIdx.x;
  int id = blockIdx.x;
  int qb = id & 63; id >>= 6;
  int hh = id & 7;  int bb = id >> 3;
  const float scale = 0.17677669529663689f;  // 1/sqrt(32)

  // Q A-tile (d=32 covers full K in one tile)
  int m_a = lane & 15;
  int khalf = ((lane >> 4) & 1) * 8;
  const _Float16* qp = Qh + (((size_t)(bb * NH + hh) * NTOK + qb * 16 + m_a) << 5) + khalf;
  v8h qlo = *(const v8h*)qp;
  v8h qhi = *(const v8h*)(qp + 16);
  v16h aq;
#pragma unroll
  for (int i = 0; i < 8; ++i) { aq[i] = qlo[i]; aq[i + 8] = qhi[i]; }

  const _Float16* kbase = Kt + ((size_t)(bb * NH + hh) * HD) * NTOK;  // head base, [d][N]
  int n_c = lane & 15;
  int mbase = (lane >> 4) * 8;

  // logits = Q*K^T * scale + cpb[rel_idx]
  for (int nk = 0; nk < 64; ++nk) {
    v16h kb;
#pragma unroll
    for (int i = 0; i < 8; ++i) {
      kb[i]     = kbase[(size_t)(khalf + i) * NTOK + nk * 16 + n_c];
      kb[i + 8] = kbase[(size_t)(khalf + 16 + i) * NTOK + nk * 16 + n_c];
    }
    v8f c = {};
    c = wmma16(aq, kb, c);
#pragma unroll
    for (int r = 0; r < 8; ++r) {
      int m = mbase + r;
      int col = nk * 16 + n_c;
      int nq = qb * 16 + m;
      int idx = relidx[(size_t)nq * NTOK + col];
      smem[m * NTOK + col] = c[r] * scale + cpb[idx * NH + hh];
    }
  }

  // softmax per row; keep 1/sum for the row this lane owns in the A layout
  float pinv = 0.0f;
  for (int m = 0; m < 16; ++m) {
    float mx = -1e30f;
    for (int t = lane; t < NTOK; t += 32) mx = fmaxf(mx, smem[m * NTOK + t]);
#pragma unroll
    for (int s = 16; s >= 1; s >>= 1) mx = fmaxf(mx, __shfl_xor(mx, s, 32));
    float sum = 0.0f;
    for (int t = lane; t < NTOK; t += 32) {
      float e = __expf(smem[m * NTOK + t] - mx);
      smem[m * NTOK + t] = e;
      sum += e;
    }
#pragma unroll
    for (int s = 16; s >= 1; s >>= 1) sum += __shfl_xor(sum, s, 32);
    float inv = 1.0f / sum;
    if ((lane & 15) == m) pinv = inv;
  }

  // O = P*V, fold 1/sum into the f16 conversion of P
  const _Float16* vbase = Vh + (((size_t)(bb * NH + hh) * NTOK) << 5);
#pragma unroll
  for (int ct = 0; ct < 2; ++ct) {
    v8f acc = {};
    for (int kc = 0; kc < 32; ++kc) {
      int k0 = kc * 32 + khalf;
      const float* sp = &smem[m_a * NTOK + k0];
      v16h a;
#pragma unroll
      for (int i = 0; i < 8; ++i) {
        a[i]     = (_Float16)(sp[i]      * pinv);
        a[i + 8] = (_Float16)(sp[i + 16] * pinv);
      }
      v16h vb;
#pragma unroll
      for (int i = 0; i < 8; ++i) {
        vb[i]     = vbase[(((size_t)(k0 + i)) << 5) + ct * 16 + n_c];
        vb[i + 8] = vbase[(((size_t)(k0 + 16 + i)) << 5) + ct * 16 + n_c];
      }
      acc = wmma16(a, vb, acc);
    }
#pragma unroll
    for (int r = 0; r < 8; ++r) {
      int m = mbase + r;
      O16[((size_t)(bb * NTOK) + qb * 16 + m) * C_ + hh * HD + ct * 16 + n_c] = (_Float16)acc[r];
    }
  }
}

// ---------------- LayerNorm over C=256, scatter to spatial [B,C,32,32] ----------------
__global__ __launch_bounds__(256)
void k_layernorm(const float* __restrict__ z, const float* __restrict__ g,
                 const float* __restrict__ b, float* __restrict__ ysp) {
  __shared__ float red[256];
  int r = blockIdx.x;   // 0..8191
  int t = threadIdx.x;
  float x = z[(size_t)r * C_ + t];
  red[t] = x; __syncthreads();
  for (int s = 128; s > 0; s >>= 1) { if (t < s) red[t] += red[t + s]; __syncthreads(); }
  float mu = red[0] * (1.0f / 256.0f); __syncthreads();
  float d = x - mu;
  red[t] = d * d; __syncthreads();
  for (int s = 128; s > 0; s >>= 1) { if (t < s) red[t] += red[t + s]; __syncthreads(); }
  float var = red[0] * (1.0f / 256.0f);
  float y = d * rsqrtf(var + 1e-5f) * g[t] + b[t];
  int bb = r >> 10, n = r & 1023, ph = n >> 5, pw = n & 31;
  ysp[((size_t)(bb * C_ + t) * PH + ph) * PW + pw] = y;
}

// ---------------- bilinear 32->128 upsample + relu + residual ----------------
__global__ void k_upsample_residual(const float* __restrict__ ysp, const float* __restrict__ xin,
                                    float* __restrict__ out) {
  size_t i = (size_t)blockIdx.x * blockDim.x + threadIdx.x;
  if (i >= (size_t)B_ * C_ * H_ * W_) return;
  __builtin_prefetch(xin + i + 8192, 0, 0);   // stream the residual input
  int j = i & 127; size_t t = i >> 7;
  int irow = t & 127; t >>= 7;
  int c = (int)(t & 255); int b = (int)(t >> 8);
  float fy = (irow + 0.5f) * 0.25f - 0.5f;
  float fx = (j + 0.5f) * 0.25f - 0.5f;
  int y0 = (int)floorf(fy), x0 = (int)floorf(fx);
  float wy = fy - y0, wx = fx - x0;
  int y0c = min(max(y0, 0), 31), y1c = min(max(y0 + 1, 0), 31);
  int x0c = min(max(x0, 0), 31), x1c = min(max(x0 + 1, 0), 31);
  const float* p = ysp + (size_t)(b * C_ + c) * PH * PW;
  float v00 = p[y0c * PW + x0c], v01 = p[y0c * PW + x1c];
  float v10 = p[y1c * PW + x0c], v11 = p[y1c * PW + x1c];
  float v = (1.0f - wy) * ((1.0f - wx) * v00 + wx * v01) + wy * ((1.0f - wx) * v10 + wx * v11);
  out[i] = xin[i] + fmaxf(v, 0.0f);
}

extern "C" void kernel_launch(void* const* d_in, const int* in_sizes, int n_in,
                              void* d_out, int out_size, void* d_ws, size_t ws_size,
                              hipStream_t stream) {
  const float* q    = (const float*)d_in[0];
  const float* kv   = (const float*)d_in[1];
  const int*   ridx = (const int*)d_in[2];
  const float* tbl  = (const float*)d_in[3];
  const float* Wq   = (const float*)d_in[4];
  const float* bq   = (const float*)d_in[5];
  const float* Wkv  = (const float*)d_in[6];
  const float* bkv  = (const float*)d_in[7];
  const float* Wo   = (const float*)d_in[8];
  const float* bo   = (const float*)d_in[9];
  const float* cw1  = (const float*)d_in[10];
  const float* cb1  = (const float*)d_in[11];
  const float* cw2  = (const float*)d_in[12];
  const float* cb2  = (const float*)d_in[13];
  const float* fw1  = (const float*)d_in[14];
  const float* fb1  = (const float*)d_in[15];
  const float* fw2  = (const float*)d_in[16];
  const float* fb2  = (const float*)d_in[17];
  const float* lng  = (const float*)d_in[18];
  const float* lnb  = (const float*)d_in[19];
  float* out = (float*)d_out;
  int L = in_sizes[3] / 2;

  char* ws = (char*)d_ws;
  size_t off = 0;
  auto carve = [&](size_t bytes) -> void* {
    void* p = ws + off;
    off = (off + bytes + 255) & ~(size_t)255;
    return p;
  };
  _Float16* wq16  = (_Float16*)carve((size_t)256 * 256 * 2);
  _Float16* wkv16 = (_Float16*)carve((size_t)256 * 512 * 2);
  _Float16* wo16  = (_Float16*)carve((size_t)256 * 256 * 2);
  _Float16* w1_16 = (_Float16*)carve((size_t)256 * FFN * 2);
  _Float16* w2_16 = (_Float16*)carve((size_t)FFN * 256 * 2);
  _Float16* qt16  = (_Float16*)carve((size_t)B_ * NTOK * C_ * 2);
  _Float16* kvt16 = (_Float16*)carve((size_t)B_ * NTOK * C_ * 2);
  _Float16* Qh16  = (_Float16*)carve((size_t)B_ * NH * NTOK * HD * 2);
  _Float16* Kt16  = (_Float16*)carve((size_t)B_ * NH * HD * NTOK * 2);
  _Float16* V16   = (_Float16*)carve((size_t)B_ * NH * NTOK * HD * 2);
  _Float16* O16   = (_Float16*)carve((size_t)B_ * NTOK * C_ * 2);
  _Float16* out16 = (_Float16*)carve((size_t)B_ * NTOK * C_ * 2);
  _Float16* hid16 = (_Float16*)carve((size_t)B_ * NTOK * FFN * 2);
  float*    zbuf  = (float*)carve((size_t)B_ * NTOK * C_ * 4);
  float*    ysp   = (float*)carve((size_t)B_ * C_ * PH * PW * 4);
  float*    cpbT  = (float*)carve((size_t)L * NH * 4);

  const int M = B_ * NTOK;   // 8192
  const int MB = M / 64;     // 128 row-blocks (4 M-tiles per block)

  // 1. weight conversion
  k_f32_to_f16<<<(256 * 256 + 255) / 256, 256, 0, stream>>>(Wq,  wq16,  256 * 256);
  k_f32_to_f16<<<(256 * 512 + 255) / 256, 256, 0, stream>>>(Wkv, wkv16, 256 * 512);
  k_f32_to_f16<<<(256 * 256 + 255) / 256, 256, 0, stream>>>(Wo,  wo16,  256 * 256);
  k_f32_to_f16<<<(256 * FFN + 255) / 256, 256, 0, stream>>>(fw1, w1_16, 256 * FFN);
  k_f32_to_f16<<<(FFN * 256 + 255) / 256, 256, 0, stream>>>(fw2, w2_16, FFN * 256);

  // 2. downsample + relu -> [B,N,C] f16
  int nds = B_ * C_ * PH * PW;
  k_downsample<<<(nds + 255) / 256, 256, 0, stream>>>(q,  qt16);
  k_downsample<<<(nds + 255) / 256, 256, 0, stream>>>(kv, kvt16);

  // 3. CPB table
  k_cpb<<<(L + 255) / 256, 256, 0, stream>>>(tbl, cw1, cb1, cw2, cb2, cpbT, L);

  // 4. Q / KV projections (WMMA, async B-panel staging)
  k_gemm16<<<MB * (256 / 16), 128, 0, stream>>>(qt16,  wq16,  bq,  M, 256, 256, EP_QSCAT,  Qh16, nullptr, nullptr);
  k_gemm16<<<MB * (512 / 16), 128, 0, stream>>>(kvt16, wkv16, bkv, M, 256, 512, EP_KVSCAT, Kt16, V16, nullptr);

  // 5. attention (WMMA + LDS softmax)
  k_attention<<<B_ * NH * (NTOK / 16), 32, 0, stream>>>(Qh16, Kt16, V16, ridx, cpbT, O16);

  // 6. Wo projection -> z (f32) + f16 copy
  k_gemm16<<<MB * (256 / 16), 128, 0, stream>>>(O16, wo16, bo, M, 256, 256, EP_F32_AND_F16, out16, nullptr, zbuf);

  // 7. FFN
  k_gemm16<<<MB * (FFN / 16), 128, 0, stream>>>(out16, w1_16, fb1, M, 256, FFN, EP_GELU_F16, hid16, nullptr, nullptr);
  k_gemm16<<<MB * (256 / 16), 128, 0, stream>>>(hid16, w2_16, fb2, M, FFN, 256, EP_ADD_F32, nullptr, nullptr, zbuf);

  // 8. LayerNorm -> spatial
  k_layernorm<<<M, 256, 0, stream>>>(zbuf, lng, lnb, ysp);

  // 9. upsample + relu + residual
  size_t ntot = (size_t)B_ * C_ * H_ * W_;
  k_upsample_residual<<<(unsigned)((ntot + 255) / 256), 256, 0, stream>>>(ysp, q, out);
}
